// LSTMFrame_19138374271632
// MI455X (gfx1250) — matrix operand order repaired and checked by reading, not compiled, not run on hardware
//
#include <hip/hip_runtime.h>
#include <hip/hip_bf16.h>

// Problem dimensions (from the reference).
#define SEQ    512
#define BATCH  64
#define DMODEL 1024
#define HID    1024
#define GATES  4096   // 4*HID rows in W
#define KDIM   2048   // DMODEL + HID
#define NWG    64     // persistent workgroups; WG g owns hidden units [16g,16g+16)
#define THREADS 256   // 8 wave32 per WG

typedef __attribute__((ext_vector_type(16))) __bf16 v16bf;
typedef __attribute__((ext_vector_type(8)))  __bf16 v8bf;
typedef __attribute__((ext_vector_type(8)))  float  v8f;
typedef __attribute__((ext_vector_type(4)))  unsigned int v4u;
typedef __attribute__((ext_vector_type(8)))  int v8i;
typedef __attribute__((ext_vector_type(4)))  int v4i;

union V16 { v16bf v; v8bf h[2]; };

#if defined(__has_builtin)
#if __has_builtin(__builtin_amdgcn_tensor_load_to_lds)
#define HAVE_TDM 1
#else
#define HAVE_TDM 0
#endif
#else
#define HAVE_TDM 0
#endif

__device__ __forceinline__ unsigned short f32_to_bf16_rne(float f) {
  unsigned u = __float_as_uint(f);
  unsigned r = u + 0x7FFFu + ((u >> 16) & 1u);
  return (unsigned short)(r >> 16);
}
// Order-preserving f32 -> u32 key for atomic min/max reductions.
__device__ __forceinline__ unsigned fkey(float f) {
  unsigned u = __float_as_uint(f);
  return (u & 0x80000000u) ? ~u : (u | 0x80000000u);
}
__device__ __forceinline__ float funkey(unsigned k) {
  return (k & 0x80000000u) ? __uint_as_float(k ^ 0x80000000u) : __uint_as_float(~k);
}

// ---------------------------------------------------------------- prep kernels

__global__ void init_kernel(unsigned* bar, unsigned* mm,
                            unsigned short* h0a, unsigned short* h0b) {
  size_t tid = (size_t)blockIdx.x * blockDim.x + threadIdx.x;
  if (tid == 0) {
    bar[0] = 0u; bar[1] = 0u;                 // count, generation
    mm[0] = 0xFFFFFFFFu; mm[1] = 0u;          // W0 min/max keys
    mm[2] = 0xFFFFFFFFu; mm[3] = 0u;          // W1 min/max keys
  }
  size_t n = (size_t)BATCH * HID;             // zero h_0 slot of both layers
  for (size_t i = tid; i < n; i += (size_t)gridDim.x * blockDim.x) {
    h0a[i] = 0; h0b[i] = 0;
  }
}

__global__ void minmax_kernel(const float* __restrict__ w, size_t n,
                              unsigned* mn, unsigned* mx) {
  __shared__ unsigned smn[THREADS], smx[THREADS];
  size_t tid = (size_t)blockIdx.x * blockDim.x + threadIdx.x;
  unsigned lmn = 0xFFFFFFFFu, lmx = 0u;
  for (size_t i = tid; i < n; i += (size_t)gridDim.x * blockDim.x) {
    unsigned k = fkey(w[i]);
    lmn = min(lmn, k); lmx = max(lmx, k);
  }
  smn[threadIdx.x] = lmn; smx[threadIdx.x] = lmx;
  __syncthreads();
  for (int s = THREADS / 2; s > 0; s >>= 1) {
    if ((int)threadIdx.x < s) {
      smn[threadIdx.x] = min(smn[threadIdx.x], smn[threadIdx.x + s]);
      smx[threadIdx.x] = max(smx[threadIdx.x], smx[threadIdx.x + s]);
    }
    __syncthreads();
  }
  if (threadIdx.x == 0) { atomicMin(mn, smn[0]); atomicMax(mx, smx[0]); }
}

// Fake-quant (8-bit affine, global scale/zp) -> bf16, matching reference forward.
__global__ void quant_kernel(const float* __restrict__ w,
                             unsigned short* __restrict__ wq,
                             size_t n, const unsigned* __restrict__ mm) {
  float wmin = fminf(funkey(mm[0]), 0.0f);
  float wmax = fmaxf(funkey(mm[1]), 0.0f);
  float scale = (wmax - wmin) * (1.0f / 255.0f);
  float zp = fminf(fmaxf(rintf(-wmin / scale), 0.0f), 255.0f);
  float clamp_low = -scale * zp;
  float clamp_high = scale * (255.0f - zp);
  size_t tid = (size_t)blockIdx.x * blockDim.x + threadIdx.x;
  for (size_t i = tid; i < n; i += (size_t)gridDim.x * blockDim.x) {
    float x = w[i];
    float q = (fminf(fmaxf(rintf(x / scale + zp), 0.0f), 255.0f) - zp) * scale;
    float cl = fminf(fmaxf(x, clamp_low), clamp_high);
    wq[i] = f32_to_bf16_rne(cl + (q - x));
  }
}

__global__ void f32_to_bf16_kernel(const float* __restrict__ in,
                                   unsigned short* __restrict__ out, size_t n) {
  size_t tid = (size_t)blockIdx.x * blockDim.x + threadIdx.x;
  for (size_t i = tid; i < n; i += (size_t)gridDim.x * blockDim.x)
    out[i] = f32_to_bf16_rne(in[i]);
}

// --------------------------------------------------- TDM weight-slice preload

#if HAVE_TDM
// Issue one 2-D Tensor-Data-Mover descriptor: tile (rows x width) of bf16,
// row stride `row_stride_elems`, global -> LDS. D# layout per ISA 8.3/8.4.
__device__ __forceinline__ void tdm_load_2d_to_lds(unsigned lds_byte_addr,
                                                   const void* gsrc,
                                                   unsigned rows,
                                                   unsigned width_elems,
                                                   unsigned row_stride_elems,
                                                   unsigned tensor_rows) {
  unsigned long long ga = (unsigned long long)(size_t)gsrc;
  v4u g0;
  g0.x = 1u;                                            // count=1 (valid, user)
  g0.y = lds_byte_addr;                                 // lds_addr
  g0.z = (unsigned)(ga & 0xFFFFFFFFu);                  // global_addr[31:0]
  g0.w = (unsigned)((ga >> 32) & 0x01FFFFFFu)           // global_addr[56:32]
       | (2u << 30);                                    // type=2 ("image")
  v8i g1;
  g1[0] = (int)(1u << 16);                              // data_size=1 -> 2 bytes
  g1[1] = (int)((width_elems & 0xFFFFu) << 16);         // tensor_dim0[15:0]
  g1[2] = (int)(((width_elems >> 16) & 0xFFFFu)         // tensor_dim0[31:16]
       | ((tensor_rows & 0xFFFFu) << 16));              // tensor_dim1[15:0]
  g1[3] = (int)(((tensor_rows >> 16) & 0xFFFFu)         // tensor_dim1[31:16]
       | ((width_elems & 0xFFFFu) << 16));              // tile_dim0
  g1[4] = (int)(rows & 0xFFFFu);                        // tile_dim1 (tile_dim2=0)
  g1[5] = (int)row_stride_elems;                        // tensor_dim0_stride[31:0]
  g1[6] = 0;                                            // stride hi / dim1_stride lo
  g1[7] = 0;                                            // dim1_stride hi
  v4i g2 = {0, 0, 0, 0};                                // 2-D tile: groups 2/3 unused
  v4i g3 = {0, 0, 0, 0};
#if __clang_major__ >= 23
  v8i g4 = {0, 0, 0, 0, 0, 0, 0, 0};
  __builtin_amdgcn_tensor_load_to_lds(g0, g1, g2, g3, g4, 0);
#else
  __builtin_amdgcn_tensor_load_to_lds(g0, g1, g2, g3, 0);
#endif
}
#endif

// ------------------------------------------------------------- LSTM main loop

// One 1024-wide K half (32 steps of k=32) of the per-wave GEMM: A(16x1024 rows of this wave's
// m-tile) x two B tiles (from LDS), accumulating two 16x16 f32 C tiles.
__device__ __forceinline__ void gemm_kslab(const __bf16* __restrict__ arow,
                                           const __bf16* __restrict__ b0base,
                                           const __bf16* __restrict__ b1base,
                                           int koff, v8f& acc0, v8f& acc1) {
#pragma unroll 4
  for (int kb = 0; kb < 32; ++kb) {
    const int k0 = kb * 32;
    V16 a, b0, b1;
    // 16-bit A/B lane layout: lanes 0-15 hold K {0..7,16..23}, lanes 16-31
    // hold K {8..15,24..31} -> two 16B runs at koff and koff+16.
    a.h[0]  = *(const v8bf*)(arow  + k0 + koff);
    a.h[1]  = *(const v8bf*)(arow  + k0 + koff + 16);
    b0.h[0] = *(const v8bf*)(b0base + k0 + koff);
    b0.h[1] = *(const v8bf*)(b0base + k0 + koff + 16);
    b1.h[0] = *(const v8bf*)(b1base + k0 + koff);
    b1.h[1] = *(const v8bf*)(b1base + k0 + koff + 16);
    acc0 = __builtin_amdgcn_wmma_f32_16x16x32_bf16(false, a.v, false, b0.v,
                                                   (short)0, acc0, false, false);
    acc1 = __builtin_amdgcn_wmma_f32_16x16x32_bf16(false, a.v, false, b1.v,
                                                   (short)0, acc1, false, false);
  }
}

__global__ __launch_bounds__(THREADS, 1)
void lstm_layer_kernel(const unsigned short* __restrict__ Xin,   // [SEQ][B][1024] bf16
                       const unsigned short* __restrict__ Wq,    // [4096][2048]   bf16
                       const float* __restrict__ bias,           // [4096]
                       unsigned short* __restrict__ Hbuf,        // [SEQ+1][B][1024] bf16, slot0 zeroed
                       float* __restrict__ y_out,                // nullable [SEQ][B][1024]
                       float* __restrict__ hN,                   // [B][1024]
                       float* __restrict__ cN,                   // [B][1024]
                       unsigned* __restrict__ bar,               // {count, generation}
                       unsigned gen_base) {
  extern __shared__ char smem[];
  __bf16* wlds = (__bf16*)smem;                                  // 64 x 2048 (256 KB)
  float*  zbuf = (float*)(smem + (size_t)64 * KDIM * 2);         // 64 x 64
  float*  cbuf = (float*)((char*)zbuf + 64 * 64 * 4);            // 64 x 16 cell state
  float*  blds = (float*)((char*)cbuf + 64 * 16 * 4);            // 64 biases

  const int tid  = threadIdx.x;
  const int g    = blockIdx.x;        // hidden-slice owner: units [16g, 16g+16)
  const int wave = tid >> 5;
  const int lane = tid & 31;
  const int mtile = wave & 3;         // batch tile
  const int n0    = (wave >> 2) * 2;  // first of two gate-column tiles

  // ---- Preload this WG's weight slice (rows {gate*1024 + 16g + j}) into LDS.
#if HAVE_TDM
  // Tensor Data Mover: one 16x2048 bf16 tile per gate block, DMA'd directly
  // global->LDS with no VGPR round-trip. TDM ignores EXEC; per-wave op, so
  // only wave 0 issues, drains TENSORcnt, then the WG syncs.
  if (wave == 0) {
#pragma unroll
    for (int gate = 0; gate < 4; ++gate) {
      const __bf16* src = (const __bf16*)Wq
          + ((size_t)gate * HID + (size_t)g * 16) * KDIM;
      unsigned lds_addr =
          (unsigned)(size_t)(wlds + (size_t)gate * 16 * KDIM);
      tdm_load_2d_to_lds(lds_addr, src, /*rows=*/16, /*width=*/KDIM,
                         /*row_stride=*/KDIM, /*tensor_rows=*/GATES);
    }
    __builtin_amdgcn_s_wait_tensorcnt(0);
  }
#else
  {
    const int cpr = KDIM / 8;                       // 256 x 16B chunks per row
    for (int idx = tid; idx < 64 * cpr; idx += THREADS) {
      int lr = idx / cpr, ck = idx % cpr;
      int gate = lr >> 4, jj = lr & 15;
      size_t grow = (size_t)(gate * HID + g * 16 + jj);
      v8bf d = *(const v8bf*)((const __bf16*)Wq + grow * KDIM + (size_t)ck * 8);
      *(v8bf*)(wlds + (size_t)lr * KDIM + (size_t)ck * 8) = d;
    }
  }
#endif
  if (tid < 64) {
    int gate = tid >> 4, jj = tid & 15;
    blds[tid] = bias[gate * HID + g * 16 + jj];
  }
  for (int i = tid; i < BATCH * 16; i += THREADS) cbuf[i] = 0.0f;
  __syncthreads();

  const int arow = mtile * 16 + (lane & 15);
  const int koff = (lane >> 4) * 8;
  const __bf16* xall = (const __bf16*)Xin;
  const __bf16* hall = (const __bf16*)Hbuf;
  const __bf16* b0base = wlds + (size_t)(n0 * 16 + (lane & 15)) * KDIM;
  const __bf16* b1base = wlds + (size_t)((n0 + 1) * 16 + (lane & 15)) * KDIM;

  for (int t = 0; t < SEQ; ++t) {
    v8f acc0 = {}; v8f acc1 = {};

    // x-half of K: no recurrence dependence — issue before the grid wait.
    const __bf16* xrow = xall + ((size_t)t * BATCH + arow) * DMODEL;
    gemm_kslab(xrow, b0base, b1base, koff, acc0, acc1);
    if (t + 1 < SEQ)
      __builtin_prefetch(xall + ((size_t)(t + 1) * BATCH + arow) * DMODEL, 0, 1);

    // Wait until every WG has published h_t (generation == gen_base + t).
    if (tid == 0) {
      unsigned target = gen_base + (unsigned)t;
      while (((volatile unsigned*)bar)[1] < target) { }
    }
    __syncthreads();
    __threadfence();

    // h-half of K.
    const __bf16* hrow = hall + ((size_t)t * BATCH + arow) * HID;
    gemm_kslab(hrow, b0base + DMODEL, b1base + DMODEL, koff, acc0, acc1);

    // Spill C tiles to LDS: VGPR r -> M = 16*mtile + (lane>=16 ? 8 : 0) + r.
    {
      int ncol = n0 * 16 + (lane & 15);
      int mrow = mtile * 16 + ((lane >> 4) ? 8 : 0);
#pragma unroll
      for (int r = 0; r < 8; ++r) {
        zbuf[(mrow + r) * 64 + ncol]      = acc0[r];
        zbuf[(mrow + r) * 64 + ncol + 16] = acc1[r];
      }
    }
    __syncthreads();

    // Gate nonlinearities + cell update for our 16 hidden units x 64 batch.
    for (int item = tid; item < BATCH * 16; item += THREADS) {
      int b = item >> 4, jj = item & 15;
      float zf = zbuf[b * 64 +      jj] + blds[     jj];
      float zi = zbuf[b * 64 + 16 + jj] + blds[16 + jj];
      float zo = zbuf[b * 64 + 32 + jj] + blds[32 + jj];
      float zu = zbuf[b * 64 + 48 + jj] + blds[48 + jj];
      float fg = 1.0f / (1.0f + __expf(-zf));
      float ig = 1.0f / (1.0f + __expf(-zi));
      float og = 1.0f / (1.0f + __expf(-zo));
      float ug = tanhf(zu);
      float c2 = ig * ug + fg * cbuf[item];
      cbuf[item] = c2;
      float h2 = og * tanhf(c2);
      size_t col = (size_t)g * 16 + jj;
      Hbuf[((size_t)(t + 1) * BATCH + b) * HID + col] = f32_to_bf16_rne(h2);
      if (y_out) y_out[((size_t)t * BATCH + b) * HID + col] = h2;
      if (t == SEQ - 1) hN[(size_t)b * HID + col] = h2;
    }
    __syncthreads();

    // Publish h_{t+1}: release fence, then sense-reversal arrival.
    if (tid == 0) {
      __threadfence();
      unsigned prev = atomicAdd(&bar[0], 1u);
      if (prev == NWG - 1) {
        bar[0] = 0;
        __threadfence();
        atomicAdd(&bar[1], 1u);   // generation -> gen_base + t + 1
      }
    }
  }

  __syncthreads();
  for (int item = tid; item < BATCH * 16; item += THREADS) {
    int b = item >> 4, jj = item & 15;
    cN[(size_t)b * HID + (size_t)g * 16 + jj] = cbuf[item];
  }
}

// ------------------------------------------------------------------ dispatch

extern "C" void kernel_launch(void* const* d_in, const int* in_sizes, int n_in,
                              void* d_out, int out_size, void* d_ws, size_t ws_size,
                              hipStream_t stream) {
  const float* x  = (const float*)d_in[0];
  const float* W0 = (const float*)d_in[1];
  const float* b0 = (const float*)d_in[2];
  const float* W1 = (const float*)d_in[3];
  const float* b1 = (const float*)d_in[4];
  (void)in_sizes; (void)n_in; (void)out_size; (void)ws_size;

  char* ws = (char*)d_ws;
  size_t off = 0;
  auto alloc = [&](size_t bytes) -> void* {
    void* p = ws + off;
    off += (bytes + 255) & ~(size_t)255;
    return p;
  };
  unsigned short* Wq0   = (unsigned short*)alloc((size_t)GATES * KDIM * 2);
  unsigned short* Wq1   = (unsigned short*)alloc((size_t)GATES * KDIM * 2);
  unsigned short* Xbf   = (unsigned short*)alloc((size_t)SEQ * BATCH * DMODEL * 2);
  unsigned short* Hbuf0 = (unsigned short*)alloc((size_t)(SEQ + 1) * BATCH * HID * 2);
  unsigned short* Hbuf1 = (unsigned short*)alloc((size_t)(SEQ + 1) * BATCH * HID * 2);
  unsigned* mm  = (unsigned*)alloc(4 * sizeof(unsigned));
  unsigned* bar = (unsigned*)alloc(2 * sizeof(unsigned));

  float* Y  = (float*)d_out;                          // y2: [512][64][1024]
  float* hN = Y + (size_t)SEQ * BATCH * HID;          // h_n: [2][64][1024]
  float* cN = hN + 2 * (size_t)BATCH * HID;           // c_n: [2][64][1024]

  const size_t nW = (size_t)GATES * KDIM;
  const size_t nX = (size_t)SEQ * BATCH * DMODEL;

  init_kernel<<<64, THREADS, 0, stream>>>(bar, mm, Hbuf0, Hbuf1);
  minmax_kernel<<<256, THREADS, 0, stream>>>(W0, nW, mm + 0, mm + 1);
  minmax_kernel<<<256, THREADS, 0, stream>>>(W1, nW, mm + 2, mm + 3);
  quant_kernel<<<512, THREADS, 0, stream>>>(W0, Wq0, nW, mm + 0);
  quant_kernel<<<512, THREADS, 0, stream>>>(W1, Wq1, nW, mm + 2);
  f32_to_bf16_kernel<<<1024, THREADS, 0, stream>>>(x, Xbf, nX);

  const size_t shmem = (size_t)64 * KDIM * 2   // weight slice
                     + (size_t)64 * 64 * 4     // z buffer
                     + (size_t)64 * 16 * 4     // cell state
                     + (size_t)64 * 4;         // biases
  hipFuncSetAttribute(reinterpret_cast<const void*>(lstm_layer_kernel),
                      hipFuncAttributeMaxDynamicSharedMemorySize, (int)shmem);

  // Layer 0: input = x(bf16); y1 kept only as bf16 recurrence buffer.
  lstm_layer_kernel<<<NWG, THREADS, shmem, stream>>>(
      Xbf, Wq0, b0, Hbuf0, nullptr, hN, cN, bar, 0u);
  // Layer 1: input = y1 = Hbuf0 slots 1..512; writes y2 + h2/c2 finals.
  lstm_layer_kernel<<<NWG, THREADS, shmem, stream>>>(
      Hbuf0 + (size_t)BATCH * HID, Wq1, b1, Hbuf1, Y,
      hN + (size_t)BATCH * HID, cN + (size_t)BATCH * HID, bar, (unsigned)SEQ);
}
